// Attention_6743098654909
// MI455X (gfx1250) — matrix operand order repaired and checked by reading, not compiled, not run on hardware
//
#include <hip/hip_runtime.h>
#include <hip/hip_bf16.h>

typedef __attribute__((ext_vector_type(16))) __bf16 v16bf;
typedef __attribute__((ext_vector_type(8)))  __bf16 v8bf;
typedef __attribute__((ext_vector_type(4)))  __bf16 v4bf;
typedef __attribute__((ext_vector_type(8)))  float  v8f;

#define NTOK  49
#define CDIM  192
#define HEADS 6
#define DH    32
#define TT    2
#define NPAD  64

__device__ __forceinline__ v16bf cat16(v8bf lo, v8bf hi) {
  return __builtin_shufflevector(lo, hi, 0, 1, 2, 3, 4, 5, 6, 7, 8, 9, 10, 11, 12, 13, 14, 15);
}

// ---------------- WMMA fragment helpers (bf16 16x16x32, wave32) ----------------
// A (16x32 bf16), row-major source: lane L holds row M=L%16;
// j=0..15 -> K=kb+j, kb=(L>=16)*16. One 32B contiguous run (two 16B loads).
__device__ __forceinline__ v16bf load_a(const __bf16* base, int ld, int row0, int col0, int lane) {
  const __bf16* p = base + (size_t)(row0 + (lane & 15)) * ld + col0 + ((lane >> 4) << 4);
  v8bf lo = *(const v8bf*)(p);
  v8bf hi = *(const v8bf*)(p + 8);
  return cat16(lo, hi);
}

// B = M^T where M is row-major (N x K): B(k,n) = M[n][k]; lane L: N=L%16, K=kb+j, kb=(L>=16)*16.
// One 32B contiguous run per lane.
__device__ __forceinline__ v16bf load_bt(const __bf16* base, int ld, int k0, int n0, int lane) {
  const __bf16* p = base + (size_t)(n0 + (lane & 15)) * ld + k0 + ((lane >> 4) << 4);
  v8bf lo = *(const v8bf*)(p);
  v8bf hi = *(const v8bf*)(p + 8);
  return cat16(lo, hi);
}

// Pre-swizzled global weight fragment: frag f stored as 32 lanes x 16 bf16 contiguous.
__device__ __forceinline__ v16bf load_bf(const __bf16* base, int frag, int lane) {
  const __bf16* p = base + (((size_t)frag * 32 + lane) << 4);
  v8bf lo = *(const v8bf*)(p);
  v8bf hi = *(const v8bf*)(p + 8);
  return cat16(lo, hi);
}

__device__ __forceinline__ v8f wmma_bf16(v16bf a, v16bf b, v8f c) {
  return __builtin_amdgcn_wmma_f32_16x16x32_bf16(false, a, false, b, (short)0, c, false, false);
}

// ---------------- weight prep ----------------
// f32 W (Cout,Cin) -> bf16 fragment-swizzled: frag (kt,nt), elem idx = ((kt*NT+nt)*32+lane)*16+j
// maps to ci = kt*32 + (lane>=16)*16 + j, co = nt*16 + lane%16.
__device__ __forceinline__ __bf16 swz(const float* __restrict__ w, int Cin, int NT, int idx) {
  int j = idx & 15;
  int lane = (idx >> 4) & 31;
  int f = idx >> 9;
  int kt = f / NT, nt = f % NT;
  int ci = kt * 32 + ((lane >> 4) << 4) + j;
  int co = nt * 16 + (lane & 15);
  return (__bf16)w[(size_t)co * Cin + ci];
}

__global__ __launch_bounds__(256) void prep_weights(const float* __restrict__ wq,
                                                    const float* __restrict__ wkv,
                                                    const float* __restrict__ wp,
                                                    __bf16* __restrict__ oq,
                                                    __bf16* __restrict__ okv,
                                                    __bf16* __restrict__ op) {
  int i = blockIdx.x * blockDim.x + threadIdx.x;
  if (i < 192 * 192) {
    oq[i] = swz(wq, 192, 12, i);
    op[i] = swz(wp, 192, 12, i);
  }
  if (i < 384 * 192) {
    okv[i] = swz(wkv, 192, 24, i);
  }
}

// ---------------- fused windowed cross-attention ----------------
// one workgroup per batch window b; inner loop over T=2 memory windows so the
// Q projection is computed once and x is read from HBM exactly once.
// 256 threads = 8 waves (wave32).
__global__ __launch_bounds__(256) void attn_kernel(
    const float* __restrict__ x, const float* __restrict__ memory,
    const float* __restrict__ b_q, const float* __restrict__ b_kv,
    const float* __restrict__ b_proj, const float* __restrict__ rpb,
    const __bf16* __restrict__ wqF, const __bf16* __restrict__ wkvF,
    const __bf16* __restrict__ wpF, float* __restrict__ out) {
  extern __shared__ char smem[];
  __bf16* sX  = (__bf16*)(smem + 0);       // 64x192 bf16 = 24576 B (dead after Q proj)
  __bf16* sM  = (__bf16*)(smem + 24576);   // 64x192 (reloaded per t)
  __bf16* sQ  = (__bf16*)(smem + 49152);   // 64x192 (pre-scaled, persists across t)
  __bf16* sK  = (__bf16*)(smem + 73728);   // 64x192
  __bf16* sVT = (__bf16*)(smem + 98304);   // 192x64 (V transposed: [c][m])
  float*  sS  = (float*)(smem + 122880);   // 6x64x64 f32 = 98304 B (scores)
  __bf16* sO  = (__bf16*)(smem + 122880);  // 64x192 bf16, overlays sS (sS dead by then)
  __bf16* sP  = (__bf16*)(smem + 0);       // 6x64x64 bf16 = 49152 B, overlays sX+sM
  float*  sB  = (float*)(smem + 221184);   // rpb table 169x6 f32

  const int tid  = threadIdx.x;
  const int lane = tid & 31;
  const int wv   = tid >> 5;
  const int b    = blockIdx.x;
  const int hi8  = (lane >> 4) << 3; // C-frag: M = r + hi8
  const float scale = 0.17677669529663687f; // 32^-0.5
  const float4 z4 = make_float4(0.f, 0.f, 0.f, 0.f);

  // ---- stage 1: load x[b] (float4), convert to bf16, pad rows to 64; stage rpb ----
  const float4* xb = (const float4*)(x + (size_t)b * NTOK * CDIM);
  for (int e = tid; e < NPAD * CDIM / 4; e += 256) {
    int r = e / (CDIM / 4);
    float4 vx = (r < NTOK) ? xb[e] : z4;
    *(v4bf*)(sX + e * 4) = (v4bf){(__bf16)vx.x, (__bf16)vx.y, (__bf16)vx.z, (__bf16)vx.w};
  }
  for (int e = tid; e < 169 * HEADS; e += 256) sB[e] = rpb[e];
  __syncthreads();

  // ---- stage 2a: Q = x@WqT + bq, pre-scaled (computed ONCE, shared by both t) ----
  for (int job = wv; job < 4 * 12; job += 8) {
    int mt = job / 12, ct = job % 12;
    v8f acc = {};
#pragma unroll
    for (int kk = 0; kk < 6; ++kk) {
      v16bf a  = load_a(sX, CDIM, mt * 16, kk * 32, lane);
      v16bf bb = load_bf(wqF, kk * 12 + ct, lane);
      acc = wmma_bf16(a, bb, acc);
    }
    int co = ct * 16 + (lane & 15);
    float bia = b_q[co];
#pragma unroll
    for (int r = 0; r < 8; ++r)
      sQ[(mt * 16 + r + hi8) * CDIM + co] = (__bf16)((acc[r] + bia) * scale);
  }
  // (no barrier needed yet: next stage's sM write region is disjoint from sQ/sX reads
  //  per-thread, and a barrier precedes any cross-thread read of sM/sQ)

  for (int t = 0; t < TT; ++t) {
    const int bt = b * TT + t;

    // ---- stage 1b: load memory[bt] tile (overwrites P-heads-3..5 region; safe:
    //      previous iteration's P reads completed before post-AV barrier) ----
    const float4* mb = (const float4*)(memory + (size_t)bt * NTOK * CDIM);
    for (int e = tid; e < NPAD * CDIM / 4; e += 256) {
      int r = e / (CDIM / 4);
      float4 vm = (r < NTOK) ? mb[e] : z4;
      *(v4bf*)(sM + e * 4) = (v4bf){(__bf16)vm.x, (__bf16)vm.y, (__bf16)vm.z, (__bf16)vm.w};
    }
    __syncthreads();

    // ---- stage 2b: [K|V] = mem@WkvT + bkv ----
    for (int job = wv; job < 4 * 24; job += 8) {
      int mt = job / 24, ctt = job % 24;
      v8f acc = {};
#pragma unroll
      for (int kk = 0; kk < 6; ++kk) {
        v16bf a  = load_a(sM, CDIM, mt * 16, kk * 32, lane);
        v16bf bb = load_bf(wkvF, kk * 24 + ctt, lane);
        acc = wmma_bf16(a, bb, acc);
      }
      int co = ctt * 16 + (lane & 15);
      float bia = b_kv[co];
      if (co < CDIM) {
#pragma unroll
        for (int r = 0; r < 8; ++r)
          sK[(mt * 16 + r + hi8) * CDIM + co] = (__bf16)(acc[r] + bia);
      } else {
        int cl = co - CDIM;
#pragma unroll
        for (int r = 0; r < 8; ++r)  // store V transposed: sVT[c][m] (contiguous in r)
          sVT[(size_t)cl * NPAD + mt * 16 + r + hi8] = (__bf16)(acc[r] + bia);
      }
    }
    __syncthreads();

    // ---- stage 3: S_h = Q_h @ K_h^T  (per head 64x64, K=32 single WMMA) ----
    for (int job = wv; job < HEADS * 16; job += 8) {
      int h = job >> 4, tt = job & 15;
      int mt = tt >> 2, nt = tt & 3;
      v16bf a  = load_a(sQ, CDIM, mt * 16, h * DH, lane);
      v16bf bb = load_bt(sK, CDIM, h * DH, nt * 16, lane);
      v8f acc = {};
      acc = wmma_bf16(a, bb, acc);
      int n = nt * 16 + (lane & 15);
#pragma unroll
      for (int r = 0; r < 8; ++r)
        sS[(h * 64 + mt * 16 + r + hi8) * 64 + n] = acc[r];
    }
    __syncthreads();

    // ---- stage 4: zero P, then bias + softmax per (head,row) -> P bf16 ----
    for (int e = tid; e < HEADS * 64 * 64 / 4; e += 256)
      *(v4bf*)(sP + e * 4) = (v4bf){(__bf16)0.f, (__bf16)0.f, (__bf16)0.f, (__bf16)0.f};
    __syncthreads();
    for (int rid = tid; rid < HEADS * NTOK; rid += 256) {
      int h = rid / NTOK, n = rid % NTOK;
      float* srow = sS + (size_t)(h * 64 + n) * 64;
      __bf16* prow = sP + (size_t)(h * 64 + n) * 64;
      int nh = n / 7, nw = n % 7;
      float mx = -1e30f;
      for (int m = 0; m < NTOK; ++m) {
        int mh = m / 7, mw = m % 7;
        int idx = (nh - mh + 6) * 13 + (nw - mw + 6);   // relative position index
        float v = srow[m] + sB[idx * HEADS + h];
        srow[m] = v;
        mx = fmaxf(mx, v);
      }
      float s = 0.0f;
      for (int m = 0; m < NTOK; ++m) {
        float e = __expf(srow[m] - mx);
        srow[m] = e;
        s += e;
      }
      float inv = 1.0f / s;
      for (int m = 0; m < NTOK; ++m) prow[m] = (__bf16)(srow[m] * inv);
    }
    __syncthreads();

    // ---- stage 5: O_h = P_h @ V_h  (per head 64x32, K=64 -> 2 WMMA) ----
    for (int job = wv; job < HEADS * 8; job += 8) {
      int h = job >> 3, tt = job & 7;
      int mt = tt >> 1, nt = tt & 1;
      v8f acc = {};
#pragma unroll
      for (int ks = 0; ks < 2; ++ks) {
        v16bf a  = load_a(sP + (size_t)h * 64 * 64, 64, mt * 16, ks * 32, lane);
        // B(k=m, n=dd) = V[m][h*32+dd] = sVT[(h*32+dd)*64 + m] -> transposed read
        v16bf bb = load_bt(sVT, NPAD, ks * 32, h * DH + nt * 16, lane);
        acc = wmma_bf16(a, bb, acc);
      }
      int col = h * DH + nt * 16 + (lane & 15);
#pragma unroll
      for (int r = 0; r < 8; ++r)
        sO[(mt * 16 + r + hi8) * CDIM + col] = (__bf16)acc[r];
    }
    __syncthreads();

    // ---- stage 6: out = O @ WpT + b_proj, write rows < 49 to global ----
    float* ob = out + (size_t)bt * NTOK * CDIM;
    for (int job = wv; job < 4 * 12; job += 8) {
      int mt = job / 12, ct = job % 12;
      v8f acc = {};
#pragma unroll
      for (int kk = 0; kk < 6; ++kk) {
        v16bf a  = load_a(sO, CDIM, mt * 16, kk * 32, lane);
        v16bf bb = load_bf(wpF, kk * 12 + ct, lane);
        acc = wmma_bf16(a, bb, acc);
      }
      int co = ct * 16 + (lane & 15);
      float bia = b_proj[co];
#pragma unroll
      for (int r = 0; r < 8; ++r) {
        int row = mt * 16 + r + hi8;
        if (row < NTOK) ob[(size_t)row * CDIM + co] = acc[r] + bia;
      }
    }
    // next iteration's stage-1b overwrites sM (= P heads 3..5) and stage-3 writes
    // sS (= sO): both are fenced by the barriers inside the next iteration before
    // any cross-thread dependence, and per-thread program order covers the rest.
  }
}

extern "C" void kernel_launch(void* const* d_in, const int* in_sizes, int n_in,
                              void* d_out, int out_size, void* d_ws, size_t ws_size,
                              hipStream_t stream) {
  const float* x      = (const float*)d_in[0];
  const float* memory = (const float*)d_in[1];
  const float* w_q    = (const float*)d_in[2];
  const float* b_q    = (const float*)d_in[3];
  const float* w_kv   = (const float*)d_in[4];
  const float* b_kv   = (const float*)d_in[5];
  const float* w_proj = (const float*)d_in[6];
  const float* b_proj = (const float*)d_in[7];
  const float* rpb    = (const float*)d_in[8];

  // bf16 fragment-swizzled weights in workspace
  __bf16* wqF  = (__bf16*)d_ws;                              // 192*192*2 = 73728 B
  __bf16* wkvF = (__bf16*)((char*)d_ws + 73728);             // 192*384*2 = 147456 B
  __bf16* wpF  = (__bf16*)((char*)d_ws + 73728 + 147456);    // 73728 B

  prep_weights<<<(384 * 192 + 255) / 256, 256, 0, stream>>>(w_q, w_kv, w_proj, wqF, wkvF, wpF);

  int B = in_sizes[0] / (NTOK * CDIM);   // 4096 batch windows
  size_t smem = 225280;                  // 220 KB of the 320 KB WGP LDS
  attn_kernel<<<B, 256, smem, stream>>>(x, memory, b_q, b_kv, b_proj, rpb,
                                        wqF, wkvF, wpF, (float*)d_out);
}
